// ScaledDotProductAttention_72799695667248
// MI455X (gfx1250) — matrix operand order repaired and checked by reading, not compiled, not run on hardware
//
#include <hip/hip_runtime.h>
#include <cstdint>
#include <cstddef>

// Problem constants (from reference): B=16, L=2048, D=256, CLAMP=0.1
#define B_   16
#define L_   2048
#define D_   256
#define TI   16      // query rows per workgroup tile
#define JC   128     // key columns staged per iteration
#define SQP  264     // sQ row stride (bf16 elems), 256 + 8 pad
#define SKP  264     // sK row stride (bf16 elems)
#define SVP  136     // sVt row stride (bf16 elems), 128 + 8 pad
#define SSP  2052    // sSlog row stride (f32 elems), 2048 + 4 pad

typedef __bf16 bf16_t;
typedef __attribute__((ext_vector_type(16))) __bf16 v16bf;
typedef __attribute__((ext_vector_type(8)))  __bf16 v8bf;
typedef __attribute__((ext_vector_type(4)))  __bf16 v4bf;
typedef __attribute__((ext_vector_type(8)))  float  v8f;

union ABFrag { v16bf v; v8bf h[2]; };

// fp32 -> bf16 (RNE): compiler selects the hardware cvt on gfx1250.
__device__ __forceinline__ bf16_t tobf(float f) {
  return static_cast<bf16_t>(f);
}

__device__ __forceinline__ v4bf tobf4(float4 f) {
  v4bf r;
  r[0] = tobf(f.x); r[1] = tobf(f.y); r[2] = tobf(f.z); r[3] = tobf(f.w);
  return r;
}

__device__ __forceinline__ v8bf cvt8_sub(const float* p, float sub) {
  v8bf r;
#pragma unroll
  for (int i = 0; i < 8; ++i) r[i] = tobf(p[i] - sub);
  return r;
}

// ln(x) for x guaranteed >= 0.1: raw v_log_f32 (base-2 TRANS op, co-issues
// with WMMA) * ln2 — skips the denorm pre-scale guard of __logf/logf.
__device__ __forceinline__ float fast_ln(float x) {
  return __builtin_amdgcn_logf(x) * 0.69314718055994530942f;
}

__device__ __forceinline__ v8f wmma_bf16(const ABFrag& a, const ABFrag& b, v8f c) {
  // D = A(16x32 bf16) * B(32x16 bf16) + C(16x16 f32)
  return __builtin_amdgcn_wmma_f32_16x16x32_bf16(
      /*neg_a=*/false, a.v, /*neg_b=*/false, b.v,
      /*c_mod=*/(short)0, c, /*reuse_a=*/false, /*reuse_b=*/false);
}

// ---------------------------------------------------------------------------
// Kernel 1: per-batch column sums of V  (vsum[b][d] = sum_j v[b,j,d])
// ---------------------------------------------------------------------------
__global__ void __launch_bounds__(256)
vsum_kernel(const float* __restrict__ v, float* __restrict__ vsum) {
  const int b = blockIdx.x;
  const int d = threadIdx.x;
  const float* p = v + (size_t)b * L_ * D_ + d;
  float s0 = 0.f, s1 = 0.f, s2 = 0.f, s3 = 0.f;
  for (int j = 0; j < L_; j += 4) {
    s0 += p[(size_t)(j + 0) * D_];
    s1 += p[(size_t)(j + 1) * D_];
    s2 += p[(size_t)(j + 2) * D_];
    s3 += p[(size_t)(j + 3) * D_];
  }
  vsum[b * D_ + d] = (s0 + s1) + (s2 + s3);
}

// ---------------------------------------------------------------------------
// Kernel 2: fused  scores -> clamp -> log -> log_softmax -> (attn, context)
//   attn[i,j]    = slog[i,j] - lse[i],      slog = log(max(q.k, 0.1))
//   context[i,d] = sum_j (slog[i,j]-4) * v[j,d]  + (4 - lse[i]) * vsum[d]
// One workgroup (8 waves) per (batch, 16-row tile).
// ---------------------------------------------------------------------------
__global__ void __launch_bounds__(256, 1)
attn_fused_kernel(const float* __restrict__ q, const float* __restrict__ k,
                  const float* __restrict__ v, const float* __restrict__ vsum,
                  float* __restrict__ ctx, float* __restrict__ attn) {
  extern __shared__ char smem_raw[];
  float*  sSlog = reinterpret_cast<float*>(smem_raw);     // TI x SSP f32
  float*  sRow  = sSlog + TI * SSP;                       // 16 f32
  float*  sLse  = sRow + 16;                              // 16 f32
  bf16_t* sQ    = reinterpret_cast<bf16_t*>(sLse + 16);   // TI x SQP bf16
  bf16_t* sK    = sQ + TI * SQP;                          // JC x SKP bf16
  bf16_t* sVt   = sK + JC * SKP;                          // D_ x SVP bf16 (transposed)

  const int b    = blockIdx.y;
  const int i0   = blockIdx.x * TI;
  const int tid  = threadIdx.x;
  const int wave = tid >> 5;
  const int lane = tid & 31;
  const int half = lane >> 4;   // 0: lanes 0-15, 1: lanes 16-31
  const int l16  = lane & 15;

  const float* qb = q + ((size_t)b * L_ + i0) * D_;
  const float* kb = k + (size_t)b * L_ * D_;
  const float* vb = v + (size_t)b * L_ * D_;
  float* attnb = attn + ((size_t)b * L_ + i0) * (size_t)L_;
  float* ctxb  = ctx  + ((size_t)b * L_ + i0) * D_;

  if (tid < 16) sRow[tid] = 0.0f;

  // Stage Q tile as bf16 (coalesced float4 reads, packed b64 LDS stores).
  for (int e = tid; e < TI * (D_ / 4); e += 256) {
    const int r = e >> 6, g = e & 63;
    const float4 f = reinterpret_cast<const float4*>(qb + (size_t)r * D_)[g];
    *reinterpret_cast<v4bf*>(sQ + r * SQP + g * 4) = tobf4(f);
  }
  __syncthreads();

  // Preload the 8 Q A-fragments (16x32 each, covering D=256).
  // A layout: lane<16 row M=lane holds K 0-7 & 16-23; lane>=16 holds K 8-15 & 24-31.
  ABFrag aq[8];
#pragma unroll
  for (int c = 0; c < 8; ++c) {
    const bf16_t* base = sQ + l16 * SQP + c * 32 + half * 8;
    aq[c].h[0] = *reinterpret_cast<const v8bf*>(base);
    aq[c].h[1] = *reinterpret_cast<const v8bf*>(base + 16);
  }

  v8f acc0 = {};  // context cols dbase..dbase+15
  v8f acc1 = {};  // context cols dbase+16..dbase+31
  float rsum[8];
#pragma unroll
  for (int c = 0; c < 8; ++c) rsum[c] = 0.0f;

  const int jw = wave * 16;  // this wave's column offset inside the JC chunk

  for (int jt = 0; jt < L_ / JC; ++jt) {
    const int j0 = jt * JC;

    // Prefetch next K/V chunk into cache (global_prefetch_b8).
    if (jt + 1 < L_ / JC) {
      const float* nk = kb + (size_t)(j0 + JC) * D_;
      const float* nv = vb + (size_t)(j0 + JC) * D_;
      for (int off = tid * 32; off < JC * D_; off += 256 * 32) {
        __builtin_prefetch(nk + off, 0, 0);
        __builtin_prefetch(nv + off, 0, 0);
      }
    }

    // Stage K rows j0..j0+JC-1 row-major bf16.
    for (int e = tid; e < JC * (D_ / 4); e += 256) {
      const int r = e >> 6, g = e & 63;
      const float4 f = reinterpret_cast<const float4*>(kb + (size_t)(j0 + r) * D_)[g];
      *reinterpret_cast<v4bf*>(sK + r * SKP + g * 4) = tobf4(f);
    }
    // Stage V rows transposed: sVt[d][jloc] (contiguous in j for B-fragments).
    for (int e = tid; e < JC * (D_ / 4); e += 256) {
      const int r = e >> 6, g = e & 63;
      const float4 f = reinterpret_cast<const float4*>(vb + (size_t)(j0 + r) * D_)[g];
      const int dd = g * 4;
      sVt[(dd + 0) * SVP + r] = tobf(f.x);
      sVt[(dd + 1) * SVP + r] = tobf(f.y);
      sVt[(dd + 2) * SVP + r] = tobf(f.z);
      sVt[(dd + 3) * SVP + r] = tobf(f.w);
    }
    __syncthreads();

    // S(16x16) = Q . K^T for this wave's columns [jw, jw+16), K-dim = D = 8x32.
    // B layout: lane<16 col N=lane holds K 0-7 & 16-23; lane>=16 holds K 8-15 & 24-31.
    v8f s = {};
#pragma unroll
    for (int c = 0; c < 8; ++c) {
      ABFrag bk;
      const bf16_t* base = sK + (jw + l16) * SKP + c * 32 + half * 8;
      bk.h[0] = *reinterpret_cast<const v8bf*>(base);
      bk.h[1] = *reinterpret_cast<const v8bf*>(base + 16);
      s = wmma_bf16(aq[c], bk, s);
    }

    // Elementwise clamp + raw v_log_f32; stash slog strip; accumulate row sums.
    // C layout: VGPR c -> row (c + 8*half), col = l16.
#pragma unroll
    for (int c = 0; c < 8; ++c) {
      const float sp = fmaxf(s[c], 0.1f);
      rsum[c] += sp;
      sSlog[(c + 8 * half) * SSP + (j0 + jw + l16)] = fast_ln(sp);
    }
    __syncthreads();

    // context += (Slog - 4) . V over this j-chunk (K-dim = JC = 4x32).
    const int dbase = wave * 32;
#pragma unroll
    for (int kc = 0; kc < 4; ++kc) {
      ABFrag as;
      const float* ab = sSlog + l16 * SSP + j0 + kc * 32 + half * 8;
      as.h[0] = cvt8_sub(ab, 4.0f);
      as.h[1] = cvt8_sub(ab + 16, 4.0f);

      const bf16_t* vp0 = sVt + (dbase + l16) * SVP + kc * 32 + half * 8;
      ABFrag b0, b1;
      b0.h[0] = *reinterpret_cast<const v8bf*>(vp0);
      b0.h[1] = *reinterpret_cast<const v8bf*>(vp0 + 16);
      const bf16_t* vp1 = vp0 + 16 * SVP;
      b1.h[0] = *reinterpret_cast<const v8bf*>(vp1);
      b1.h[1] = *reinterpret_cast<const v8bf*>(vp1 + 16);

      acc0 = wmma_bf16(as, b0, acc0);
      acc1 = wmma_bf16(as, b1, acc1);
    }
    __syncthreads();
  }

  // Row sums -> lse = log(sum_j max(score, 0.1))  (exact log_softmax denom).
#pragma unroll
  for (int c = 0; c < 8; ++c) atomicAdd(&sRow[c + 8 * half], rsum[c]);
  __syncthreads();
  if (tid < 16) sLse[tid] = fast_ln(sRow[tid]);
  __syncthreads();

  // Context writeout: ctx = acc + (4 - lse) * vsum.
  {
    const int dbase = wave * 32;
    const float vs0 = vsum[b * D_ + dbase + l16];
    const float vs1 = vsum[b * D_ + dbase + 16 + l16];
#pragma unroll
    for (int c = 0; c < 8; ++c) {
      const int m = c + 8 * half;
      const float f = 4.0f - sLse[m];
      ctxb[(size_t)m * D_ + dbase + l16]      = acc0[c] + f * vs0;
      ctxb[(size_t)m * D_ + dbase + 16 + l16] = acc1[c] + f * vs1;
    }
  }

  // Attention writeout (single pass, vectorized): attn = slog - lse.
  for (int e = tid; e < TI * (L_ / 4); e += 256) {
    const int r = e >> 9, g = e & 511;
    const float lse = sLse[r];
    float4 t = *reinterpret_cast<const float4*>(sSlog + r * SSP + g * 4);
    t.x -= lse; t.y -= lse; t.z -= lse; t.w -= lse;
    reinterpret_cast<float4*>(attnb + (size_t)r * L_)[g] = t;
  }
}

// ---------------------------------------------------------------------------
extern "C" void kernel_launch(void* const* d_in, const int* in_sizes, int n_in,
                              void* d_out, int out_size, void* d_ws, size_t ws_size,
                              hipStream_t stream) {
  (void)in_sizes; (void)n_in; (void)out_size; (void)ws_size;
  const float* q = (const float*)d_in[0];
  const float* k = (const float*)d_in[1];
  const float* v = (const float*)d_in[2];

  float* ctx  = (float*)d_out;                       // (B, L, D)
  float* attn = ctx + (size_t)B_ * L_ * D_;          // (B, L, L)
  float* vsum = (float*)d_ws;                        // (B, D) = 16 KB scratch

  constexpr size_t SMEM =
      (size_t)(TI * SSP + 32) * sizeof(float) +
      (size_t)(TI * SQP + JC * SKP + D_ * SVP) * sizeof(bf16_t);  // ~277 KB

  (void)hipFuncSetAttribute(reinterpret_cast<const void*>(attn_fused_kernel),
                            hipFuncAttributeMaxDynamicSharedMemorySize,
                            (int)SMEM);

  vsum_kernel<<<dim3(B_), dim3(D_), 0, stream>>>(v, vsum);
  attn_fused_kernel<<<dim3(L_ / TI, B_), dim3(256), SMEM, stream>>>(
      q, k, v, vsum, ctx, attn);
}
